// BahdanauAttention_88811333747112
// MI455X (gfx1250) — matrix operand order repaired and checked by reading, not compiled, not run on hardware
//
#include <hip/hip_runtime.h>

// ---------------------------------------------------------------------------
// Bahdanau attention for MI455X (gfx1250, wave32, WMMA).
//   B=32, S=2048, ENC=DEC=1024
//   ws: [ dec_t (32x1024 f32) | scores (32x2048 f32) ]  = 384 KB
//   d_out: [ context (32x1024) | attn (32x2048) ] f32
// ---------------------------------------------------------------------------

#define BB 32
#define SS 2048
#define DD 1024

typedef __attribute__((ext_vector_type(16))) __bf16 v16bf;
typedef __attribute__((ext_vector_type(8)))  __bf16 v8bf;
typedef __attribute__((ext_vector_type(4)))  __bf16 v4bf;
typedef __attribute__((ext_vector_type(8)))  float  v8f;

// ---- main scores kernel tiling ----
#define MB 128                 // rows (M) per workgroup, always within one batch
#define NP 128                 // N columns per pass
#define A_STRIDE 1032          // (DD + 8) bf16  -> conflict-free ds_load_b128
#define B_STRIDE 40            // (32 + 8) bf16  -> 16B-aligned, low-conflict
#define B_BUF_ELTS (NP * B_STRIDE)
#define LDS_A_BYTES (MB * A_STRIDE * 2)        // 264192
#define LDS_B_BYTES (2 * B_BUF_ELTS * 2)       // 20480 (double-buffered)
#define LDS_S_BYTES (MB * 4)                   // 512
#define LDS_TOTAL   (LDS_A_BYTES + LDS_B_BYTES + LDS_S_BYTES)  // 285184 < 320KB

static __device__ inline v16bf make_v16(v8bf lo, v8bf hi) {
  v16bf r;
#pragma unroll
  for (int i = 0; i < 8; ++i) { r[i] = lo[i]; r[i + 8] = hi[i]; }
  return r;
}

// Stage W_enc[k0..k0+31][nbase..nbase+127] transposed -> Bbuf[n][k] (bf16).
// float4 global reads along N (fully coalesced), 4 scattered b16 LDS stores.
// Also prefetch the following K-chunk (gfx1250 global_prefetch path).
static __device__ inline void stage_B(__bf16* __restrict__ Bbuf,
                                      const float* __restrict__ Wenc,
                                      int k0, int nbase, int tid) {
#pragma unroll
  for (int j = 0; j < 4; ++j) {
    const int idx = tid + j * 256;       // 0..1023 float4 slots (32 k x 32 n4)
    const int kk = idx >> 5;             // 0..31
    const int n4 = idx & 31;             // float4 index within 128-wide row
    const float* src = Wenc + (long)(k0 + kk) * DD + nbase + n4 * 4;
    const float4 f = *reinterpret_cast<const float4*>(src);
    if (k0 + 32 < DD) __builtin_prefetch(src + 32 * DD, 0, 0);
    __bf16* d = Bbuf + (n4 * 4) * B_STRIDE + kk;
    d[0 * B_STRIDE] = (__bf16)f.x;
    d[1 * B_STRIDE] = (__bf16)f.y;
    d[2 * B_STRIDE] = (__bf16)f.z;
    d[3 * B_STRIDE] = (__bf16)f.w;
  }
}

// ---------------------------------------------------------------------------
// Kernel 1: dec_t = decoder_hidden @ W_dec + b_dec   (32 x 1024), bf16 WMMA.
// 128 waves total: wave -> (mt in 0..1, nt in 0..63), one 16x16 tile each.
// ---------------------------------------------------------------------------
__global__ __launch_bounds__(256) void decproj_kernel(
    const float* __restrict__ dh, const float* __restrict__ Wdec,
    const float* __restrict__ bdec, float* __restrict__ dect)
{
  const int gw   = blockIdx.x * 8 + (threadIdx.x >> 5);  // 0..127
  const int lane = threadIdx.x & 31;
  const int half = lane >> 4, lr = lane & 15;
  const int mt = gw >> 6;      // 0..1
  const int nt = gw & 63;      // 0..63
  const int row = mt * 16 + lr;   // < 32
  const int col = nt * 16 + lr;

  v8f acc = {};
  for (int k0 = 0; k0 < DD; k0 += 32) {
    const int kA0 = k0 + (half ? 16 : 0);
    const int kA1 = k0 + (half ? 24 : 8);
    v16bf a, bm;
    const float* ap0 = dh + row * DD + kA0;
    const float* ap1 = dh + row * DD + kA1;
#pragma unroll
    for (int i = 0; i < 8; ++i) { a[i] = (__bf16)ap0[i]; a[i + 8] = (__bf16)ap1[i]; }
    const int kb = k0 + (half ? 16 : 0);
#pragma unroll
    for (int i = 0; i < 16; ++i) bm[i] = (__bf16)Wdec[(kb + i) * DD + col];
    acc = __builtin_amdgcn_wmma_f32_16x16x32_bf16(false, a, false, bm,
                                                  (short)0, acc, false, false);
  }
  const float bbias = bdec[col];
#pragma unroll
  for (int r = 0; r < 8; ++r) {
    const int m = mt * 16 + half * 8 + r;
    dect[m * DD + col] = acc[r] + bbias;
  }
}

// ---------------------------------------------------------------------------
// Kernel 2: fused  scores[b,s] = v . tanh(enc[b,s,:]@W_enc + b_enc + dec_t[b]) + b_v
// One WG = 128 rows. A block (128x1024 bf16) resident in LDS for the whole
// kernel (CDNA5 320KB LDS); W_enc K-chunks double-buffered in LDS (one
// barrier per K-step). 8 waves as 2(M)x4(N), 4x2 register tiles each ->
// 8 v_wmma per K-step with A-fragment reuse across the N register tiles.
// ---------------------------------------------------------------------------
__global__ __launch_bounds__(256) void scores_kernel(
    const float* __restrict__ enc, const float* __restrict__ Wenc,
    const float* __restrict__ benc, const float* __restrict__ dect,
    const float* __restrict__ vv, const float* __restrict__ bv,
    float* __restrict__ scores)
{
  extern __shared__ char lds[];
  __bf16* Ash  = (__bf16*)lds;                               // [MB][A_STRIDE]
  __bf16* Bsh  = (__bf16*)(lds + LDS_A_BYTES);               // 2 x [NP][B_STRIDE]
  float*  Sred = (float*)(lds + LDS_A_BYTES + LDS_B_BYTES);  // [MB]

  const int tid  = threadIdx.x;
  const int wave = tid >> 5, lane = tid & 31;
  const int half = lane >> 4, lr = lane & 15;
  const int wm = wave >> 2;   // 0..1  -> M-tiles wm*4 .. wm*4+3
  const int wn = wave & 3;    // 0..3  -> 32 N columns per pass
  const long row0 = (long)blockIdx.x * MB;
  const int  b    = (int)(row0 / SS);      // 128 | 2048 -> single batch per WG

  // ---- Stage entire A block (128 x 1024 f32 -> bf16) into LDS; HBM read once
  {
    const float* src = enc + row0 * DD;
    for (int i = tid; i < (MB * DD) / 4; i += 256) {
      const int r  = i >> 8;        // DD/4 = 256 float4 per row
      const int c4 = i & 255;
      const float4 f = reinterpret_cast<const float4*>(src + (long)r * DD)[c4];
      v4bf t; t[0] = (__bf16)f.x; t[1] = (__bf16)f.y; t[2] = (__bf16)f.z; t[3] = (__bf16)f.w;
      *reinterpret_cast<v4bf*>(Ash + r * A_STRIDE + c4 * 4) = t;
    }
  }
  if (tid < MB) Sred[tid] = 0.0f;

  float part[32];
#pragma unroll
  for (int i = 0; i < 32; ++i) part[i] = 0.0f;

  for (int p = 0; p < DD / NP; ++p) {          // 8 passes over N
    const int nbase = p * NP;
    v8f acc[4][2];
#pragma unroll
    for (int m = 0; m < 4; ++m)
#pragma unroll
      for (int n = 0; n < 2; ++n) { v8f z = {}; acc[m][n] = z; }

    if (p == 0) { /* A-stage barrier folds into the first B barrier */ }
    stage_B(Bsh, Wenc, 0, nbase, tid);
    __syncthreads();

    for (int ks = 0; ks < DD / 32; ++ks) {
      const int k0 = ks * 32;
      const __bf16* cur = Bsh + (ks & 1) * B_BUF_ELTS;

      // stage next K-chunk into the other buffer (no readers yet)
      if (ks + 1 < DD / 32)
        stage_B(Bsh + ((ks + 1) & 1) * B_BUF_ELTS, Wenc, k0 + 32, nbase, tid);

      // A fragments (contiguous K: lane<16 K{0..15}, lane>=16 K{16..31})
      const int kA0 = k0 + (half ? 16 : 0);
      const int kA1 = k0 + (half ? 24 : 8);
      v16bf af[4];
#pragma unroll
      for (int m = 0; m < 4; ++m) {
        const int r = wm * 64 + m * 16 + lr;
        const v8bf lo = *reinterpret_cast<const v8bf*>(Ash + r * A_STRIDE + kA0);
        const v8bf hi = *reinterpret_cast<const v8bf*>(Ash + r * A_STRIDE + kA1);
        af[m] = make_v16(lo, hi);
      }
      // B fragments (lane<16: K 0..15 of column lr; lane>=16: K 16..31) + WMMAs
#pragma unroll
      for (int n = 0; n < 2; ++n) {
        const int ncol = wn * 32 + n * 16 + lr;    // column within pass
        const __bf16* bp = cur + ncol * B_STRIDE + (half ? 16 : 0);
        const v8bf lo = *reinterpret_cast<const v8bf*>(bp);
        const v8bf hi = *reinterpret_cast<const v8bf*>(bp + 8);
        const v16bf bf = make_v16(lo, hi);
#pragma unroll
        for (int m = 0; m < 4; ++m)
          acc[m][n] = __builtin_amdgcn_wmma_f32_16x16x32_bf16(
              false, af[m], false, bf, (short)0, acc[m][n], false, false);
      }
      __syncthreads();   // next-chunk staged AND this chunk's reads complete
    }

    // epilogue for this pass: bias + dec_t + tanh, dot with v (enc_t never hits HBM)
#pragma unroll
    for (int n = 0; n < 2; ++n) {
      const int ncol = nbase + wn * 32 + n * 16 + lr;   // global column
      const float bias = benc[ncol] + dect[b * DD + ncol];
      const float vn   = vv[ncol];
#pragma unroll
      for (int m = 0; m < 4; ++m)
#pragma unroll
        for (int r = 0; r < 8; ++r)
          part[m * 8 + r] += tanhf(acc[m][n][r] + bias) * vn;
    }
  }

  // reduce the 16 column-lanes of each half-wave, then across waves via LDS
#pragma unroll
  for (int i = 0; i < 32; ++i) {
    float s = part[i];
    s += __shfl_xor(s, 1, 32);
    s += __shfl_xor(s, 2, 32);
    s += __shfl_xor(s, 4, 32);
    s += __shfl_xor(s, 8, 32);
    part[i] = s;
  }
  if (lr == 0) {
#pragma unroll
    for (int m = 0; m < 4; ++m)
#pragma unroll
      for (int r = 0; r < 8; ++r)
        atomicAdd(&Sred[wm * 64 + m * 16 + half * 8 + r], part[m * 8 + r]);
  }
  __syncthreads();
  if (tid < MB) scores[row0 + tid] = Sred[tid] + bv[0];
}

// ---------------------------------------------------------------------------
// Kernel 3: softmax over S per batch row.
// ---------------------------------------------------------------------------
__global__ __launch_bounds__(256) void softmax_kernel(
    const float* __restrict__ scores, float* __restrict__ attn)
{
  __shared__ float red[256];
  const int bidx = blockIdx.x, tid = threadIdx.x;
  const float* s = scores + (long)bidx * SS;
  float* a = attn + (long)bidx * SS;

  float m = -3.4e38f;
  for (int i = tid; i < SS; i += 256) m = fmaxf(m, s[i]);
  red[tid] = m; __syncthreads();
  for (int off = 128; off > 0; off >>= 1) {
    if (tid < off) red[tid] = fmaxf(red[tid], red[tid + off]);
    __syncthreads();
  }
  const float mx = red[0];
  __syncthreads();

  float sum = 0.0f;
  for (int i = tid; i < SS; i += 256) {
    const float e = expf(s[i] - mx);
    a[i] = e; sum += e;
  }
  red[tid] = sum; __syncthreads();
  for (int off = 128; off > 0; off >>= 1) {
    if (tid < off) red[tid] += red[tid + off];
    __syncthreads();
  }
  const float inv = 1.0f / red[0];
  for (int i = tid; i < SS; i += 256) a[i] *= inv;
}

// ---------------------------------------------------------------------------
// Kernel 4: context[b,e] = sum_s attn[b,s] * enc[b,s,e]
// Pure 256MB HBM streaming pass (~11us floor at 23.3 TB/s). attn row staged
// in LDS; float4 columns -> 2KB fully-coalesced read per block per s-step.
// ---------------------------------------------------------------------------
__global__ __launch_bounds__(128) void context_kernel(
    const float* __restrict__ enc, const float* __restrict__ attn,
    float* __restrict__ ctx)
{
  __shared__ float a[SS];
  const int bidx = blockIdx.y;
  const int e4 = blockIdx.x * 128 + threadIdx.x;   // float4 column, 0..255
  for (int s = threadIdx.x; s < SS; s += 128) a[s] = attn[(long)bidx * SS + s];
  __syncthreads();

  const float4* ep = reinterpret_cast<const float4*>(enc + (long)bidx * SS * DD) + e4;
  float4 c0 = {0.f, 0.f, 0.f, 0.f}, c1 = {0.f, 0.f, 0.f, 0.f};
  for (int s = 0; s < SS; s += 2) {
    const float w0 = a[s], w1 = a[s + 1];
    const float4 x0 = ep[(long)s * (DD / 4)];
    const float4 x1 = ep[(long)(s + 1) * (DD / 4)];
    c0.x += w0 * x0.x; c0.y += w0 * x0.y; c0.z += w0 * x0.z; c0.w += w0 * x0.w;
    c1.x += w1 * x1.x; c1.y += w1 * x1.y; c1.z += w1 * x1.z; c1.w += w1 * x1.w;
  }
  float4 out;
  out.x = c0.x + c1.x; out.y = c0.y + c1.y; out.z = c0.z + c1.z; out.w = c0.w + c1.w;
  reinterpret_cast<float4*>(ctx)[bidx * (DD / 4) + e4] = out;
}

// ---------------------------------------------------------------------------
extern "C" void kernel_launch(void* const* d_in, const int* in_sizes, int n_in,
                              void* d_out, int out_size, void* d_ws, size_t ws_size,
                              hipStream_t stream) {
  (void)in_sizes; (void)n_in; (void)out_size; (void)ws_size;
  const float* dh   = (const float*)d_in[0];  // decoder_hidden (32,1024)
  const float* enc  = (const float*)d_in[1];  // encoder_outputs (32,2048,1024)
  const float* Wenc = (const float*)d_in[2];  // (1024,1024)
  const float* benc = (const float*)d_in[3];  // (1024)
  const float* Wdec = (const float*)d_in[4];  // (1024,1024)
  const float* bdec = (const float*)d_in[5];  // (1024)
  const float* vv   = (const float*)d_in[6];  // (1024)
  const float* bv   = (const float*)d_in[7];  // scalar

  float* ctx    = (float*)d_out;              // (32,1024)
  float* attn   = (float*)d_out + BB * DD;    // (32,2048)
  float* dect   = (float*)d_ws;               // (32,1024)
  float* scores = (float*)d_ws + BB * DD;     // (32,2048)

  (void)hipFuncSetAttribute((const void*)scores_kernel,
                            hipFuncAttributeMaxDynamicSharedMemorySize, LDS_TOTAL);

  decproj_kernel<<<16, 256, 0, stream>>>(dh, Wdec, bdec, dect);
  scores_kernel<<<(BB * SS) / MB, 256, LDS_TOTAL, stream>>>(
      enc, Wenc, benc, dect, vv, bv, scores);
  softmax_kernel<<<BB, 256, 0, stream>>>(scores, attn);
  context_kernel<<<dim3(DD / 512, BB), 128, 0, stream>>>(enc, attn, ctx);
}
